// LIF_LSTMCell_14370960572434
// MI455X (gfx1250) — matrix operand order repaired and checked by reading, not compiled, not run on hardware
//
#include <hip/hip_runtime.h>
#include <hip/hip_bf16.h>

typedef __bf16 v16bf __attribute__((ext_vector_type(16)));
typedef __bf16 v8bf  __attribute__((ext_vector_type(8)));
typedef __bf16 v4bf  __attribute__((ext_vector_type(4)));
typedef float  v8f   __attribute__((ext_vector_type(8)));
typedef int    v4i   __attribute__((vector_size(16)));   // builtin's expected element type

#define LIF_THRESH 0.3f

constexpr int B_   = 16384;
constexpr int IND  = 1024;
constexpr int OUTD = 1024;
constexpr int BM   = 128;   // rows per workgroup
constexpr int BN   = 32;    // outd columns per workgroup (x4 gates)
constexpr int KT   = 32;    // k per step (WMMA bf16 K)
constexpr int NKT  = IND / KT;
constexpr int PITCH = 40;   // LDS row pitch in bf16 (80B, 16B aligned, spreads banks)

#if __has_builtin(__builtin_amdgcn_global_load_async_to_lds_b128) && \
    __has_builtin(__builtin_amdgcn_s_wait_asynccnt)
#define USE_ASYNC_LDS 1
#else
#define USE_ASYNC_LDS 0
#endif

typedef __attribute__((address_space(3))) char* lds_ptr_t;
typedef __attribute__((address_space(3))) v4i*  lds_v4i_t;

// ---------------- pre-pass: f32 -> bf16 ----------------
__global__ void f32_to_bf16_kernel(const float* __restrict__ in,
                                   __bf16* __restrict__ out, int n) {
  int i = (blockIdx.x * blockDim.x + threadIdx.x) * 4;
  if (i < n) {
    float4 f = *(const float4*)(in + i);
    v4bf o;
    o[0] = (__bf16)f.x; o[1] = (__bf16)f.y; o[2] = (__bf16)f.z; o[3] = (__bf16)f.w;
    *(v4bf*)(out + i) = o;
  }
}

__global__ void bias_sum_kernel(const float* __restrict__ bh,
                                const float* __restrict__ bi,
                                float* __restrict__ bsum, int n) {
  int i = blockIdx.x * blockDim.x + threadIdx.x;
  if (i < n) bsum[i] = bh[i] + bi[i];
}

// ---------------- fused dual-GEMM + LIF-LSTM epilogue ----------------
__global__ __launch_bounds__(256)
void lif_lstm_kernel(const __bf16* __restrict__ Xb, const __bf16* __restrict__ Hb,
                     const __bf16* __restrict__ Whb, const __bf16* __restrict__ Wib,
                     const float* __restrict__ bsum, const float* __restrict__ cx,
                     float* __restrict__ hy, float* __restrict__ cy) {
  __shared__ __bf16 sH[2][BM][PITCH];
  __shared__ __bf16 sX[2][BM][PITCH];
  __shared__ __bf16 sWh[2][4][BN][PITCH];
  __shared__ __bf16 sWi[2][4][BN][PITCH];

  const int t    = threadIdx.x;
  const int wave = t >> 5;
  const int lane = t & 31;
  const int mw   = wave >> 1;          // 0..3 : 32-row strip
  const int nw   = wave & 1;           // 0..1 : 16-col strip
  const int mBase = blockIdx.y * BM;
  const int nBase = blockIdx.x * BN;

  v8f acc[4][2] = {};                  // [gate][mfrag] 16x16 f32 tiles

  // ---- per-thread staging geometry: 256 threads x 2 chunks of 16B per tensor ----
  // activations: slot s -> row = s>>2, kq = s&3
  // weights:     slot s -> g = s>>7, n = (s>>2)&31, kq = s&3
  const __bf16* gp[4][2];              // [tensor H,X,Wh,Wi][chunk]
  lds_ptr_t     lp[4][2];              // LDS target in buffer 0
#pragma unroll
  for (int i = 0; i < 2; ++i) {
    int s = t + i * 256;
    int arow = s >> 2, akq = s & 3;
    int g = s >> 7, wn = (s >> 2) & 31, wkq = s & 3;
    size_t aoff = (size_t)(mBase + arow) * IND + akq * 8;
    size_t woff = (size_t)(g * OUTD + nBase + wn) * IND + wkq * 8;
    gp[0][i] = Hb + aoff;
    gp[1][i] = Xb + aoff;
    gp[2][i] = Whb + woff;
    gp[3][i] = Wib + woff;
    lp[0][i] = (lds_ptr_t)&sH[0][arow][akq * 8];
    lp[1][i] = (lds_ptr_t)&sX[0][arow][akq * 8];
    lp[2][i] = (lds_ptr_t)&sWh[0][g][wn][wkq * 8];
    lp[3][i] = (lds_ptr_t)&sWi[0][g][wn][wkq * 8];
  }
  const int bstride[4] = {BM * PITCH * 2, BM * PITCH * 2,
                          4 * BN * PITCH * 2, 4 * BN * PITCH * 2};

#if USE_ASYNC_LDS
  // direct global -> LDS async copy; advances global pointers by one k-step
  auto stage = [&](int buf) {
#pragma unroll
    for (int i = 0; i < 2; ++i) {
#pragma unroll
      for (int j = 0; j < 4; ++j) {
        __builtin_amdgcn_global_load_async_to_lds_b128(
            (v4i*)gp[j][i],
            (lds_v4i_t)(lp[j][i] + (buf ? bstride[j] : 0)),
            0, 0);
        gp[j][i] += KT;
      }
    }
  };
#else
  // fallback: synchronous global -> reg -> LDS
  auto stage = [&](int buf) {
#pragma unroll
    for (int i = 0; i < 2; ++i) {
#pragma unroll
      for (int j = 0; j < 4; ++j) {
        uint4 r = *(const uint4*)gp[j][i];
        *(uint4*)(lp[j][i] + (buf ? bstride[j] : 0)) = r;
        gp[j][i] += KT;
      }
    }
  };
#endif

  // per-lane fragment geometry (CDNA5 16-bit WMMA layouts)
  const int mrow = lane & 15;
  const int kbA  = (lane >> 4) * 8;    // A: K = kbA..kbA+7 and kbA+16..kbA+23
  const int kbB  = (lane >> 4) * 16;   // unused; B loads use kbA, same K split as A
  const int nn   = nw * 16 + (lane & 15);

  auto compute = [&](int buf) {
    v16bf ah[2], ax[2];
#pragma unroll
    for (int m = 0; m < 2; ++m) {
      int r = mw * 32 + m * 16 + mrow;
      v8bf lo = *(const v8bf*)&sH[buf][r][kbA];
      v8bf hi = *(const v8bf*)&sH[buf][r][kbA + 16];
#pragma unroll
      for (int i = 0; i < 8; ++i) { ah[m][i] = lo[i]; ah[m][8 + i] = hi[i]; }
      lo = *(const v8bf*)&sX[buf][r][kbA];
      hi = *(const v8bf*)&sX[buf][r][kbA + 16];
#pragma unroll
      for (int i = 0; i < 8; ++i) { ax[m][i] = lo[i]; ax[m][8 + i] = hi[i]; }
    }
#pragma unroll
    for (int g = 0; g < 4; ++g) {
      v16bf bwh, bwi;
      {
        v8bf lo = *(const v8bf*)&sWh[buf][g][nn][kbA];
        v8bf hi = *(const v8bf*)&sWh[buf][g][nn][kbA + 16];
#pragma unroll
        for (int i = 0; i < 8; ++i) { bwh[i] = lo[i]; bwh[8 + i] = hi[i]; }
        lo = *(const v8bf*)&sWi[buf][g][nn][kbA];
        hi = *(const v8bf*)&sWi[buf][g][nn][kbA + 16];
#pragma unroll
        for (int i = 0; i < 8; ++i) { bwi[i] = lo[i]; bwi[8 + i] = hi[i]; }
      }
#pragma unroll
      for (int m = 0; m < 2; ++m) {
        acc[g][m] = __builtin_amdgcn_wmma_f32_16x16x32_bf16(
            false, ah[m], false, bwh, (short)0, acc[g][m], false, false);
        acc[g][m] = __builtin_amdgcn_wmma_f32_16x16x32_bf16(
            false, ax[m], false, bwi, (short)0, acc[g][m], false, false);
      }
    }
  };

  // ---- double-buffered main loop ----
  stage(0);
#if USE_ASYNC_LDS
  __builtin_amdgcn_s_wait_asynccnt(0);
#endif
  __syncthreads();
  for (int kt = 0; kt < NKT; ++kt) {
    int buf = kt & 1;
    if (kt + 1 < NKT) stage(buf ^ 1);   // async loads overlap the WMMAs below
    compute(buf);
    if (kt + 1 < NKT) {
#if USE_ASYNC_LDS
      __builtin_amdgcn_s_wait_asynccnt(0);
#endif
      __syncthreads();
    }
  }

  // ---- LIF-LSTM epilogue (all in registers) ----
  const int col = nBase + nw * 16 + (lane & 15);
  float bias[4];
#pragma unroll
  for (int g = 0; g < 4; ++g) bias[g] = bsum[g * OUTD + col];

#pragma unroll
  for (int m = 0; m < 2; ++m) {
#pragma unroll
    for (int r = 0; r < 8; ++r) {
      int row = mBase + mw * 32 + m * 16 + r + 8 * (lane >> 4);
      size_t off = (size_t)row * OUTD + col;
      float ig = acc[0][m][r] + bias[0];
      float fg = acc[1][m][r] + bias[1];
      float cg = acc[2][m][r] + bias[2];
      float og = acc[3][m][r] + bias[3];
      float ip = (ig >= LIF_THRESH) ? 1.0f : 0.0f;
      float fp = (fg >= LIF_THRESH) ? 1.0f : 0.0f;
      float cp = (cg >= LIF_THRESH) ? 1.0f : 0.0f;
      float op = (og >= LIF_THRESH) ? 1.0f : 0.0f;
      float cyv = fp * cx[off] + ip * cp;
      float cpost = (cyv >= LIF_THRESH) ? 1.0f : 0.0f;
      hy[off] = op * cpost;
      cy[off] = cyv;
    }
  }
}

// ---------------- launch ----------------
extern "C" void kernel_launch(void* const* d_in, const int* in_sizes, int n_in,
                              void* d_out, int out_size, void* d_ws, size_t ws_size,
                              hipStream_t stream) {
  const float* x   = (const float*)d_in[0];  // [16384,1024]
  const float* hx  = (const float*)d_in[1];  // [16384,1024]
  const float* cx  = (const float*)d_in[2];  // [16384,1024]
  const float* Wh  = (const float*)d_in[3];  // [4096,1024]
  const float* bh  = (const float*)d_in[4];  // [4096]
  const float* Wi  = (const float*)d_in[5];  // [4096,1024]
  const float* bi  = (const float*)d_in[6];  // [4096]

  char* ws = (char*)d_ws;
  const size_t ACT = (size_t)B_ * IND;        // 16M elements
  const size_t WSZ = (size_t)4 * OUTD * IND;  // 4M elements
  __bf16* Xb   = (__bf16*)(ws);
  __bf16* Hb   = (__bf16*)(ws + ACT * 2);
  __bf16* Whb  = (__bf16*)(ws + ACT * 4);
  __bf16* Wib  = (__bf16*)(ws + ACT * 4 + WSZ * 2);
  float*  bsum = (float* )(ws + ACT * 4 + WSZ * 4);

  const int CT = 256;
  f32_to_bf16_kernel<<<(int)(ACT / (CT * 4)), CT, 0, stream>>>(x,  Xb,  (int)ACT);
  f32_to_bf16_kernel<<<(int)(ACT / (CT * 4)), CT, 0, stream>>>(hx, Hb,  (int)ACT);
  f32_to_bf16_kernel<<<(int)(WSZ / (CT * 4)), CT, 0, stream>>>(Wh, Whb, (int)WSZ);
  f32_to_bf16_kernel<<<(int)(WSZ / (CT * 4)), CT, 0, stream>>>(Wi, Wib, (int)WSZ);
  bias_sum_kernel<<<(4 * OUTD) / CT, CT, 0, stream>>>(bh, bi, bsum, 4 * OUTD);

  float* hy = (float*)d_out;
  float* cy = (float*)d_out + (size_t)B_ * OUTD;
  dim3 grid(OUTD / BN, B_ / BM);   // 32 x 128
  lif_lstm_kernel<<<grid, 256, 0, stream>>>(Xb, Hb, Whb, Wib, bsum, cx, hy, cy);
}
